// SelfAttention_996432413375
// MI455X (gfx1250) — compile-verified, matrix operand order checked by reading, not run on hardware
//
#include <hip/hip_runtime.h>
#include <hip/hip_bf16.h>
#include <stdint.h>

// ---------------- problem constants ----------------
#define BATCH    2
#define SEQ      4096
#define DIM      2048
#define NHEADS   16
#define HEAD_DIM 128
#define TDIM     (3 * DIM)        // 6144
#define MROWS    (BATCH * SEQ)    // 8192
#define EPS      1e-6f

typedef __attribute__((ext_vector_type(16))) __bf16 v16bf;
typedef __attribute__((ext_vector_type(8)))  float  v8f;

union FragU {
    v16bf          v;
    unsigned short us[16];
    unsigned int   ui[8];
};

// float -> bf16 (round to nearest even, NaN-agnostic)
__device__ __forceinline__ unsigned short f2bf(float f) {
    unsigned int u = __float_as_uint(f);
    unsigned int r = u + 0x7fffu + ((u >> 16) & 1u);
    return (unsigned short)(r >> 16);
}

// K index of first bf16 in VGPR i for this lane-half
// (CDNA5 16-bit A-matrix 16x32 layout; also used for B fragments loaded
//  from a [Ncol, K] row-major buffer)
__device__ __forceinline__ int kmap(int i, int half) {
    return (i < 4) ? (half * 8 + i * 2) : (16 + half * 8 + (i - 4) * 2);
}

// Load one 16x32 bf16 fragment from row-major [rows, ld] (elements).
// Works for global or LDS-resident buffers (ui[0..3] and ui[4..7] are
// each 16B contiguous -> b128 loads).
__device__ __forceinline__ v16bf load_frag(const unsigned short* base, int ld,
                                           int rowBase, int kOff) {
    int lane = threadIdx.x & 31;
    int m    = lane & 15;
    int half = lane >> 4;
    FragU f;
#pragma unroll
    for (int i = 0; i < 8; ++i) {
        int k0  = kmap(i, half);
        size_t off = (size_t)(rowBase + m) * (size_t)ld + (size_t)(kOff + k0);
        f.ui[i] = *reinterpret_cast<const unsigned int*>(base + off);
    }
    return f.v;
}

// Async 16-byte copy global -> LDS (tracked by ASYNCcnt).
__device__ __forceinline__ void async_cp_b128(const unsigned short* gsrc,
                                              unsigned short* lds_dst) {
    unsigned int lds_off = (unsigned int)(uintptr_t)lds_dst;  // addr[31:0] = LDS address
    asm volatile("global_load_async_to_lds_b128 %0, %1, off"
                 :
                 : "v"(lds_off), "v"(gsrc)
                 : "memory");
}

__device__ __forceinline__ void async_wait0() {
    asm volatile("s_wait_asynccnt 0" ::: "memory");
}

// ---------------- kernel 0a: x -> bf16 ----------------
__global__ void cvt_x_kernel(const float* __restrict__ x,
                             unsigned short* __restrict__ xb, int n) {
    int i = blockIdx.x * blockDim.x + threadIdx.x;
    if (i < n) xb[i] = f2bf(x[i]);
}

// ---------------- kernel 0b: W -> bf16 transposed [TDIM, DIM] ----------------
__global__ void cvt_wt_kernel(const float* __restrict__ w,
                              unsigned short* __restrict__ wt, int n) {
    int i = blockIdx.x * blockDim.x + threadIdx.x;
    if (i < n) {
        int k = i / TDIM;
        int c = i % TDIM;
        wt[(size_t)c * DIM + k] = f2bf(w[i]);
    }
}

// ---------------- kernel 1: qkv = x @ W + b -------------------------------
// Block tile 64(M) x 256(N); 256 threads = 8 waves in 2(M) x 4(N) layout,
// each wave computes 32x64 (8 WMMA tiles). A/B K-slices async-staged in LDS.
// grid: (TDIM/256, MROWS/64)
__global__ void qkv_gemm_kernel(const unsigned short* __restrict__ xb,
                                const unsigned short* __restrict__ wt,
                                const float* __restrict__ bias,
                                float* __restrict__ qkv) {
    __shared__ __align__(16) unsigned short Atile[64 * 32];   // 4 KiB
    __shared__ __align__(16) unsigned short Btile[256 * 32];  // 16 KiB

    const int tid  = threadIdx.x;
    const int wave = tid >> 5;
    const int lane = tid & 31;
    const int half = lane >> 4;
    const int m16  = lane & 15;
    const int wm   = wave & 1;   // 0..1
    const int wn   = wave >> 1;  // 0..3

    const int blockM = blockIdx.y * 64;
    const int blockN = blockIdx.x * 256;

    v8f acc[2][4] = {};

    for (int kk = 0; kk < DIM; kk += 32) {
        __syncthreads();  // prior iteration's LDS reads done before overwrite
        // A: 64 rows x 64B = 256 chunks; one per thread
        {
            int row = tid >> 2, ch = tid & 3;
            async_cp_b128(xb + (size_t)(blockM + row) * DIM + kk + ch * 8,
                          Atile + row * 32 + ch * 8);
        }
        // B: 256 rows x 64B = 1024 chunks; four per thread
#pragma unroll
        for (int r = 0; r < 4; ++r) {
            int chunk = tid + r * 256;
            int row = chunk >> 2, ch = chunk & 3;
            async_cp_b128(wt + (size_t)(blockN + row) * DIM + kk + ch * 8,
                          Btile + row * 32 + ch * 8);
        }
        async_wait0();
        __syncthreads();  // tiles visible to all waves

        v16bf af[2], bf[4];
#pragma unroll
        for (int s = 0; s < 2; ++s) af[s] = load_frag(Atile, 32, wm * 32 + s * 16, 0);
#pragma unroll
        for (int t = 0; t < 4; ++t) bf[t] = load_frag(Btile, 32, wn * 64 + t * 16, 0);
#pragma unroll
        for (int s = 0; s < 2; ++s)
#pragma unroll
            for (int t = 0; t < 4; ++t)
                acc[s][t] = __builtin_amdgcn_wmma_f32_16x16x32_bf16(
                    false, af[s], false, bf[t], (short)0, acc[s][t], false, false);
    }

#pragma unroll
    for (int t = 0; t < 4; ++t) {
        const int   col = blockN + wn * 64 + t * 16 + m16;
        const float bv  = bias[col];
#pragma unroll
        for (int s = 0; s < 2; ++s)
#pragma unroll
            for (int i = 0; i < 8; ++i) {
                int row = blockM + wm * 32 + s * 16 + i + 8 * half;
                qkv[(size_t)row * TDIM + col] = acc[s][t][i] + bv;
            }
    }
}

// ---------------- kernel 2: RMSNorm + RoPE + pack Q/K bf16, V^T bf16 --------
// grid: (SEQ, NHEADS, BATCH), block: 128 (one thread per d)
__global__ void normrope_kernel(const float* __restrict__ qkv,
                                const float* __restrict__ rot,
                                const float* __restrict__ qn_w,
                                const float* __restrict__ kn_w,
                                unsigned short* __restrict__ Qb,
                                unsigned short* __restrict__ Kb,
                                unsigned short* __restrict__ Vt) {
    __shared__ float sq[HEAD_DIM], sk[HEAD_DIM];
    __shared__ float redq[HEAD_DIM], redk[HEAD_DIM];

    const int d = threadIdx.x;
    const int n = blockIdx.x, h = blockIdx.y, b = blockIdx.z;

    const size_t base = ((size_t)(b * SEQ + n)) * TDIM + (size_t)(h * HEAD_DIM + d) * 3;
    const float q = qkv[base + 0];
    const float k = qkv[base + 1];
    const float v = qkv[base + 2];

    redq[d] = q * q;
    redk[d] = k * k;
    __syncthreads();
#pragma unroll
    for (int s = 64; s > 0; s >>= 1) {
        if (d < s) { redq[d] += redq[d + s]; redk[d] += redk[d + s]; }
        __syncthreads();
    }
    const float rq = rsqrtf(redq[0] * (1.0f / HEAD_DIM) + EPS);
    const float rk = rsqrtf(redk[0] * (1.0f / HEAD_DIM) + EPS);
    __syncthreads();

    sq[d] = q * rq * qn_w[d];
    sk[d] = k * rk * kn_w[d];
    __syncthreads();

    // rot layout: (N, d/2, 2, 2); out_r = rot[r][0]*t0 + rot[r][1]*t1
    const int    i  = d >> 1, r = d & 1;
    const size_t rb = (((size_t)n * (HEAD_DIM / 2) + i) * 2 + r) * 2;
    const float  r0 = rot[rb + 0], r1 = rot[rb + 1];
    const float  qo = r0 * sq[2 * i] + r1 * sq[2 * i + 1];
    const float  ko = r0 * sk[2 * i] + r1 * sk[2 * i + 1];

    const size_t bh   = (size_t)(b * NHEADS + h);
    const size_t rowO = (bh * SEQ + n) * HEAD_DIM + d;
    Qb[rowO] = f2bf(qo);
    Kb[rowO] = f2bf(ko);
    Vt[(bh * HEAD_DIM + d) * SEQ + n] = f2bf(v);
}

// ---------------- kernel 3: flash attention (WMMA + online softmax) ---------
// grid: (SEQ/64, NHEADS, BATCH), block: 128 (4 waves, 16 q-rows each).
// K/V tiles for each 32-key step async-staged in LDS, shared by all waves.
__global__ void attn_kernel(const unsigned short* __restrict__ Qb,
                            const unsigned short* __restrict__ Kb,
                            const unsigned short* __restrict__ Vt,
                            float* __restrict__ out) {
    __shared__ __align__(16) unsigned short Ktile[32 * 128];  // 8 KiB
    __shared__ __align__(16) unsigned short Vtile[128 * 32];  // 8 KiB
    __shared__ float ldsP[4 * 16 * 32];                       // 8 KiB

    const int tid  = threadIdx.x;
    const int wave = tid >> 5;
    const int lane = tid & 31;
    const int half = lane >> 4;
    const int m16  = lane & 15;
    const int h    = blockIdx.y, b = blockIdx.z;
    const int qrow = blockIdx.x * 64 + wave * 16;

    const size_t bh = (size_t)(b * NHEADS + h);
    const unsigned short* Qh = Qb + bh * SEQ * HEAD_DIM;
    const unsigned short* Kh = Kb + bh * SEQ * HEAD_DIM;
    const unsigned short* Vh = Vt + bh * HEAD_DIM * SEQ;

    const float scale = 0.08838834764831845f;  // 1/sqrt(128)

    // Q fragments (from global, once)
    v16bf aQ[4];
#pragma unroll
    for (int c = 0; c < 4; ++c) aQ[c] = load_frag(Qh, HEAD_DIM, qrow, c * 32);

    float mrow[8], lrow[8];
    v8f   O[8];
#pragma unroll
    for (int i = 0; i < 8; ++i) { mrow[i] = -1e30f; lrow[i] = 0.0f; O[i] = (v8f){}; }

    float* myP = ldsP + wave * (16 * 32);

    for (int jt = 0; jt < SEQ; jt += 32) {
        __syncthreads();  // prior iteration's tile reads complete
        // K tile: 32 rows x 256B = 512 chunks; 4 per thread
#pragma unroll
        for (int r = 0; r < 4; ++r) {
            int chunk = tid + r * 128;
            int row = chunk >> 4, ch = chunk & 15;
            async_cp_b128(Kh + (size_t)(jt + row) * HEAD_DIM + ch * 8,
                          Ktile + row * HEAD_DIM + ch * 8);
        }
        // V^T tile: 128 rows x 64B = 512 chunks; 4 per thread
#pragma unroll
        for (int r = 0; r < 4; ++r) {
            int chunk = tid + r * 128;
            int row = chunk >> 2, ch = chunk & 3;
            async_cp_b128(Vh + (size_t)row * SEQ + jt + ch * 8,
                          Vtile + row * 32 + ch * 8);
        }
        async_wait0();
        __syncthreads();  // tiles ready

        // S = Q K^T for 32 keys: two 16x16 tiles (from LDS)
        v8f S0 = {}, S1 = {};
#pragma unroll
        for (int c = 0; c < 4; ++c) {
            v16bf b0 = load_frag(Ktile, HEAD_DIM, 0,  c * 32);
            v16bf b1 = load_frag(Ktile, HEAD_DIM, 16, c * 32);
            S0 = __builtin_amdgcn_wmma_f32_16x16x32_bf16(
                false, aQ[c], false, b0, (short)0, S0, false, false);
            S1 = __builtin_amdgcn_wmma_f32_16x16x32_bf16(
                false, aQ[c], false, b1, (short)0, S1, false, false);
        }

        // online softmax per row (row = i + 8*half, spread over 16 lanes)
#pragma unroll
        for (int i = 0; i < 8; ++i) {
            float s0 = S0[i] * scale;
            float s1 = S1[i] * scale;
            float rmax = fmaxf(s0, s1);
#pragma unroll
            for (int mk = 1; mk < 16; mk <<= 1)
                rmax = fmaxf(rmax, __shfl_xor(rmax, mk, 32));
            float mnew  = fmaxf(mrow[i], rmax);
            float alpha = __expf(mrow[i] - mnew);
            float p0 = __expf(s0 - mnew);
            float p1 = __expf(s1 - mnew);
            float rsum = p0 + p1;
#pragma unroll
            for (int mk = 1; mk < 16; mk <<= 1)
                rsum += __shfl_xor(rsum, mk, 32);
            lrow[i] = lrow[i] * alpha + rsum;
            mrow[i] = mnew;

            int row = i + 8 * half;
            myP[row * 32 + m16]      = p0;
            myP[row * 32 + 16 + m16] = p1;
#pragma unroll
            for (int t = 0; t < 8; ++t) O[t][i] *= alpha;
        }
        __syncthreads();  // P staged (also orders within wave)

        // re-layout P (C layout -> A fragment, f32 -> bf16) via LDS
        FragU fp;
#pragma unroll
        for (int i = 0; i < 8; ++i) {
            int k0 = kmap(i, half);
            fp.us[2 * i + 0] = f2bf(myP[m16 * 32 + k0 + 0]);
            fp.us[2 * i + 1] = f2bf(myP[m16 * 32 + k0 + 1]);
        }

        // O += P * V (V^T tile [d, key] in LDS -> contiguous key pairs)
#pragma unroll
        for (int t = 0; t < 8; ++t) {
            v16bf bv = load_frag(Vtile, 32, t * 16, 0);
            O[t] = __builtin_amdgcn_wmma_f32_16x16x32_bf16(
                false, fp.v, false, bv, (short)0, O[t], false, false);
        }
    }

    // final 1/l scale and store to (b, n, h*128 + d) fp32
#pragma unroll
    for (int i = 0; i < 8; ++i) {
        float inv = 1.0f / lrow[i];
        int   row = qrow + i + 8 * half;
#pragma unroll
        for (int t = 0; t < 8; ++t) {
            int col = h * HEAD_DIM + t * 16 + m16;
            out[((size_t)(b * SEQ + row)) * DIM + col] = O[t][i] * inv;
        }
    }
}

// ---------------- launcher ----------------
extern "C" void kernel_launch(void* const* d_in, const int* in_sizes, int n_in,
                              void* d_out, int out_size, void* d_ws, size_t ws_size,
                              hipStream_t stream) {
    const float* x    = (const float*)d_in[0];
    const float* rot  = (const float*)d_in[1];
    const float* Wq   = (const float*)d_in[2];
    const float* bq   = (const float*)d_in[3];
    const float* qn_w = (const float*)d_in[4];
    const float* kn_w = (const float*)d_in[5];
    float*       out  = (float*)d_out;

    // workspace carve-up (bytes)
    char* ws = (char*)d_ws;
    unsigned short* xb  = (unsigned short*)(ws);                       // 32 MiB
    unsigned short* wt  = (unsigned short*)(ws + (size_t)33554432);    // 24 MiB
    float*          qkv = (float*)(ws + (size_t)58720256);             // 192 MiB
    unsigned short* Qb  = (unsigned short*)(ws + (size_t)260046848);   // 32 MiB
    unsigned short* Kb  = (unsigned short*)(ws + (size_t)293601280);   // 32 MiB
    unsigned short* Vt  = (unsigned short*)(ws + (size_t)327155712);   // 32 MiB

    // 0) fp32 -> bf16 staging
    {
        int n = MROWS * DIM;
        cvt_x_kernel<<<(n + 255) / 256, 256, 0, stream>>>(x, xb, n);
        int nw = DIM * TDIM;
        cvt_wt_kernel<<<(nw + 255) / 256, 256, 0, stream>>>(Wq, wt, nw);
    }
    // 1) QKV GEMM
    {
        dim3 grid(TDIM / 256, MROWS / 64);
        qkv_gemm_kernel<<<grid, 256, 0, stream>>>(xb, wt, bq, qkv);
    }
    // 2) RMSNorm + RoPE + pack
    {
        dim3 grid(SEQ, NHEADS, BATCH);
        normrope_kernel<<<grid, HEAD_DIM, 0, stream>>>(qkv, rot, qn_w, kn_w,
                                                       Qb, Kb, Vt);
    }
    // 3) attention
    {
        dim3 grid(SEQ / 64, NHEADS, BATCH);
        attn_kernel<<<grid, 128, 0, stream>>>(Qb, Kb, Vt, out);
    }
}
